// FrequencySparseAttention_24567212933752
// MI455X (gfx1250) — matrix operand order repaired and checked by reading, not compiled, not run on hardware
//
#include <hip/hip_runtime.h>
#include <hip/hip_bf16.h>

// ---------------- problem constants ----------------
#define BB   2
#define LL   2048
#define DD   1024
#define HH   16
#define DH   64
#define TAU  0.1f

typedef __bf16 bf16_t;
typedef __attribute__((ext_vector_type(16))) __bf16 v16bf;
typedef __attribute__((ext_vector_type(8)))  float  v8f;
typedef __attribute__((ext_vector_type(4)))  unsigned int v4u;
typedef __attribute__((ext_vector_type(8)))  int v8i;
typedef __attribute__((ext_vector_type(4)))  int v4i;

#if __has_builtin(__builtin_amdgcn_tensor_load_to_lds)
#define HAVE_TDM 1
#else
#define HAVE_TDM 0
#endif

// WMMA wrapper: D = A(16x32 bf16) * B(32x16 bf16) + C(16x16 f32)
__device__ __forceinline__ v8f wmma_bf16(v16bf a, v16bf b, v8f c) {
    return __builtin_amdgcn_wmma_f32_16x16x32_bf16(
        /*neg_a=*/false, a, /*neg_b=*/false, b,
        /*c_mod=*/(short)0, c, /*reuse_a=*/false, /*reuse_b=*/false);
}

__device__ __forceinline__ bf16_t to_bf16(float x)  { return (bf16_t)x; }
__device__ __forceinline__ bf16_t to_bf16(bf16_t x) { return x; }

#if HAVE_TDM
// ---------------------------------------------------------------------------
// Tensor Data Mover: 2-D tile load Global -> LDS (packed rows in LDS).
// D# packing per CDNA5 ISA ch.8.3/8.4:
//   group0: [1:0]=count(1), [63:32]=lds_addr, [120:64]=global_addr, [127:126]=type(2)
//   group1: [17:16]=data_size, [79:48]=tensor_dim0, [111:80]=tensor_dim1,
//           [127:112]=tile_dim0, [143:128]=tile_dim1, [207:160]=tensor_dim0_stride
// ---------------------------------------------------------------------------
__device__ __forceinline__ void tdm_load_2d(unsigned lds_addr,
                                            unsigned long long gaddr,
                                            unsigned dsize_code,   // 0=1B 1=2B 2=4B
                                            unsigned tile_d0,      // k elems per row
                                            unsigned tile_d1,      // rows
                                            unsigned tensor_d0,    // tensor k extent
                                            unsigned tensor_d1,    // tensor row extent
                                            unsigned long long stride0) {
    v4u g0;
    g0[0] = 1u;                                               // count=1, load
    g0[1] = lds_addr;
    g0[2] = (unsigned)(gaddr & 0xffffffffull);
    g0[3] = (unsigned)((gaddr >> 32) & 0x1ffffffull) | 0x80000000u;  // type=2
    v8i g1;
    g1[0] = (int)(dsize_code << 16);                          // data_size; mask=0
    g1[1] = (int)((tensor_d0 & 0xffffu) << 16);               // tensor_dim0 lo16
    g1[2] = (int)((tensor_d0 >> 16) | ((tensor_d1 & 0xffffu) << 16));
    g1[3] = (int)((tensor_d1 >> 16) | (tile_d0 << 16));       // tile_dim0
    g1[4] = (int)(tile_d1 & 0xffffu);                         // tile_dim1, tile_dim2=0
    g1[5] = (int)(stride0 & 0xffffffffull);                   // dim0 stride lo32
    g1[6] = (int)((stride0 >> 32) & 0xffffull);               // dim0 stride hi16
    g1[7] = 0;
    v4i gz = {0, 0, 0, 0};
#if defined(__clang_major__) && __clang_major__ >= 23
    v8i gz8 = {0, 0, 0, 0, 0, 0, 0, 0};
    __builtin_amdgcn_tensor_load_to_lds(g0, g1, gz, gz, gz8, 0);
#else
    __builtin_amdgcn_tensor_load_to_lds(g0, g1, gz, gz, 0);
#endif
}
__device__ __forceinline__ unsigned lds_off(const void* p) {
    // low 32 bits of a generic pointer into LDS = LDS byte address
    return (unsigned)(unsigned long long)p;
}
#endif

// ---------------------------------------------------------------------------
// Kernel 1: frequency gate.  gate[b,s] = sigmoid((mean|ec[b,s,:]| - tau)*10)
// ---------------------------------------------------------------------------
__global__ void __launch_bounds__(256) gate_kernel(const float* __restrict__ ec,
                                                   float* __restrict__ gate) {
    int wave = (blockIdx.x * blockDim.x + threadIdx.x) >> 5;
    int lane = threadIdx.x & 31;
    if (wave >= BB * LL) return;
    const float* row = ec + (size_t)wave * DD;
    float sum = 0.f;
    #pragma unroll 8
    for (int i = lane; i < DD; i += 32) sum += fabsf(row[i]);
    #pragma unroll
    for (int off = 16; off > 0; off >>= 1) sum += __shfl_xor(sum, off, 32);
    float mean = sum * (1.0f / (float)DD);
    float g = 1.0f / (1.0f + __expf(-(mean - TAU) * 10.0f));
    if (lane == 0) gate[wave] = g;
}

// ---------------------------------------------------------------------------
// One K-step of the wave's 16(m) x 64(n) tile from staged LDS blocks.
// ---------------------------------------------------------------------------
template <typename XT>
__device__ __forceinline__ void gemm_tile_step(const XT* __restrict__ As_,
                                               const float* __restrict__ Bs_,
                                               int wid, int half, int lr,
                                               v8f acc[4]) {
    // A fragment: 16-bit A 16x32 layout (contiguous: K=half*16+e)
    v16bf a;
    const XT* arow = &As_[(wid * 16 + lr) * 32];
    #pragma unroll
    for (int e = 0; e < 8; ++e) a[e]     = to_bf16(arow[half * 16 + e]);
    #pragma unroll
    for (int e = 0; e < 8; ++e) a[8 + e] = to_bf16(arow[half * 16 + 8 + e]);

    #pragma unroll
    for (int dt = 0; dt < 4; ++dt) {
        v16bf b;
        const float* brow = &Bs_[(dt * 16 + lr) * 32];
        #pragma unroll
        for (int e = 0; e < 16; ++e) b[e] = (bf16_t)brow[half * 16 + e];
        acc[dt] = wmma_bf16(a, b, acc[dt]);
    }
}

// ---------------------------------------------------------------------------
// Linear layer  Y = X @ W^T + b  -- double-buffered TDM->LDS + bf16 WMMA.
// Workgroup (8 waves) computes 128(m) x 64(n); wave w owns rows w*16..w*16+15
// and all 64 cols (4 accumulators).  K loop: 32/step, ping-pong LDS buffers:
//   issue TDM for k+32 into buf[nxt]  ||  WMMA consumes buf[cur]
//   then s_wait_tensorcnt 0 + barrier, swap.
// ---------------------------------------------------------------------------
template <typename XT, bool TO_HEADS>
__global__ void __launch_bounds__(256) linear_wmma(const XT* __restrict__ X,
                                                   const float* __restrict__ W,
                                                   const float* __restrict__ bias,
                                                   void* __restrict__ Yv) {
    __shared__ XT    As[2][128 * 32];  // A block: 128 rows x 32 k  (x2 buffers)
    __shared__ float Bs[2][64 * 32];   // B block: 64 rows (n) x 32 k

    const int NB = DD / 64;                       // 16 col blocks
    int tm = (blockIdx.x / NB) * 128;
    int tn = (blockIdx.x % NB) * 64;
    int wid  = threadIdx.x >> 5;
    int lane = threadIdx.x & 31;
    int half = lane >> 4;
    int lr   = lane & 15;

    v8f acc[4] = {};

#if HAVE_TDM
    // prologue: stage first K-tile
    if (wid == 0) {
        tdm_load_2d(lds_off(&As[0][0]),
                    (unsigned long long)(const void*)(X + (size_t)tm * DD),
                    sizeof(XT) == 4 ? 2u : 1u, 32u, 128u,
                    (unsigned)DD, (unsigned)(BB * LL), (unsigned long long)DD);
        tdm_load_2d(lds_off(&Bs[0][0]),
                    (unsigned long long)(const void*)(W + (size_t)tn * DD),
                    2u, 32u, 64u,
                    (unsigned)DD, (unsigned)DD, (unsigned long long)DD);
        __builtin_amdgcn_s_wait_tensorcnt(0);
    }
    __syncthreads();

    int cur = 0;
    for (int k0 = 0; k0 < DD; k0 += 32) {
        int nxt = cur ^ 1;
        bool more = (k0 + 32) < DD;
        if (more && wid == 0) {           // async: TDM streams next tile
            tdm_load_2d(lds_off(&As[nxt][0]),
                        (unsigned long long)(const void*)(X + (size_t)tm * DD + k0 + 32),
                        sizeof(XT) == 4 ? 2u : 1u, 32u, 128u,
                        (unsigned)DD, (unsigned)(BB * LL), (unsigned long long)DD);
            tdm_load_2d(lds_off(&Bs[nxt][0]),
                        (unsigned long long)(const void*)(W + (size_t)tn * DD + k0 + 32),
                        2u, 32u, 64u,
                        (unsigned)DD, (unsigned)DD, (unsigned long long)DD);
        }
        gemm_tile_step<XT>(&As[cur][0], &Bs[cur][0], wid, half, lr, acc);
        __syncthreads();                  // everyone done reading buf[cur]
        if (more) {
            if (wid == 0) __builtin_amdgcn_s_wait_tensorcnt(0);
            __syncthreads();              // buf[nxt] complete & visible
            cur = nxt;
        }
    }
#else
    // cooperative fallback: single-buffered staging with plain loads
    for (int k0 = 0; k0 < DD; k0 += 32) {
        for (int i = threadIdx.x; i < 128 * 32; i += 256) {
            int r = i >> 5, c = i & 31;
            As[0][i] = X[(size_t)(tm + r) * DD + k0 + c];
        }
        for (int i = threadIdx.x; i < 64 * 32; i += 256) {
            int r = i >> 5, c = i & 31;
            Bs[0][i] = W[(size_t)(tn + r) * DD + k0 + c];
        }
        __syncthreads();
        gemm_tile_step<XT>(&As[0][0], &Bs[0][0], wid, half, lr, acc);
        __syncthreads();
    }
#endif

    // epilogue: C layout -> row = r + 8*half (within wave's 16), col = lr
    #pragma unroll
    for (int dt = 0; dt < 4; ++dt) {
        int n = tn + dt * 16 + lr;
        float bn = bias[n];
        #pragma unroll
        for (int r = 0; r < 8; ++r) {
            int m = tm + wid * 16 + r + 8 * half;
            float y = acc[dt][r] + bn;
            if constexpr (TO_HEADS) {
                int b_ = m / LL, l_ = m % LL;
                int h_ = n / DH, d_ = n % DH;
                ((bf16_t*)Yv)[(((size_t)b_ * HH + h_) * LL + l_) * DH + d_] = (bf16_t)y;
            } else {
                ((float*)Yv)[(size_t)m * DD + n] = y;
            }
        }
    }
}

// ---------------------------------------------------------------------------
// Kernel 3: flash attention with per-key frequency gate.
// One wave handles 16 query rows of one (b,h); sweeps S in steps of 32 keys.
// ---------------------------------------------------------------------------
__global__ void __launch_bounds__(256) attn_kernel(const bf16_t* __restrict__ Q,
                                                   const bf16_t* __restrict__ Km,
                                                   const bf16_t* __restrict__ V,
                                                   const float*  __restrict__ gate,
                                                   bf16_t* __restrict__ att) {
    __shared__ bf16_t pbuf[8][16 * 32];                      // per-wave P staging
    int wid  = threadIdx.x >> 5;
    int lane = threadIdx.x & 31;
    int half = lane >> 4;
    int lr   = lane & 15;

    int gwave   = blockIdx.x * 8 + wid;                      // 0..4095
    int qblocks = LL / 16;                                   // 128
    int bh = gwave / qblocks;
    int qb = gwave % qblocks;
    int b_ = bh / HH;
    int h_ = bh % HH;

    const bf16_t* Qbase = Q  + (size_t)bh * LL * DH;
    const bf16_t* Kbase = Km + (size_t)bh * LL * DH;
    const bf16_t* Vbase = V  + (size_t)bh * LL * DH;
    const float*  grow  = gate + (size_t)b_ * LL;

    // Resident Q A-fragments for DH=64: two K-steps of 32
    v16bf qa[2];
    const bf16_t* qptr = Qbase + (size_t)(qb * 16 + lr) * DH;
    #pragma unroll
    for (int t = 0; t < 2; ++t) {
        int k0 = t * 32;
        #pragma unroll
        for (int e = 0; e < 8; ++e) qa[t][e]     = qptr[k0 + half * 16 + e];
        #pragma unroll
        for (int e = 0; e < 8; ++e) qa[t][8 + e] = qptr[k0 + half * 16 + 8 + e];
    }

    float mrow[8], lrow[8];
    #pragma unroll
    for (int r = 0; r < 8; ++r) { mrow[r] = -3.0e38f; lrow[r] = 0.f; }
    v8f acc[4] = {};

    const float scale = 0.125f;                              // 1/sqrt(64)

    for (int s0 = 0; s0 < LL; s0 += 32) {
        v8f sc[2] = {};
        #pragma unroll
        for (int t = 0; t < 2; ++t) {
            const bf16_t* kptr = Kbase + (size_t)(s0 + t * 16 + lr) * DH;
            if (s0 + 32 < LL)
                __builtin_prefetch(kptr + 32 * DH, 0, 1);    // global_prefetch
            #pragma unroll
            for (int k0 = 0; k0 < DH; k0 += 32) {
                v16bf bfrag = *(const v16bf*)(kptr + k0 + half * 16);
                sc[t] = wmma_bf16(qa[k0 / 32], bfrag, sc[t]);
            }
        }

        float g0 = grow[s0 + lr];
        float g1 = grow[s0 + 16 + lr];

        #pragma unroll
        for (int r = 0; r < 8; ++r) {
            float v0 = sc[0][r] * scale;
            float v1 = sc[1][r] * scale;
            float mx = fmaxf(v0, v1);
            #pragma unroll
            for (int off = 8; off > 0; off >>= 1) mx = fmaxf(mx, __shfl_xor(mx, off, 32));
            float newm = fmaxf(mrow[r], mx);
            float corr = __expf(mrow[r] - newm);
            float e0 = __expf(v0 - newm);
            float e1 = __expf(v1 - newm);
            float rs = e0 + e1;
            #pragma unroll
            for (int off = 8; off > 0; off >>= 1) rs += __shfl_xor(rs, off, 32);
            lrow[r] = lrow[r] * corr + rs;
            mrow[r] = newm;
            int prow = r + 8 * half;
            pbuf[wid][prow * 32 + lr]      = (bf16_t)(e0 * g0);   // gate post-softmax
            pbuf[wid][prow * 32 + 16 + lr] = (bf16_t)(e1 * g1);
            #pragma unroll
            for (int dt = 0; dt < 4; ++dt) acc[dt][r] *= corr;
        }
        __builtin_amdgcn_wave_barrier();

        v16bf pa;
        #pragma unroll
        for (int e = 0; e < 8; ++e) pa[e]     = pbuf[wid][lr * 32 + half * 16 + e];
        #pragma unroll
        for (int e = 0; e < 8; ++e) pa[8 + e] = pbuf[wid][lr * 32 + half * 16 + 8 + e];
        __builtin_amdgcn_wave_barrier();

        #pragma unroll
        for (int dt = 0; dt < 4; ++dt) {
            v16bf vb;
            int d = dt * 16 + lr;
            #pragma unroll
            for (int e = 0; e < 16; ++e)
                vb[e] = Vbase[(size_t)(s0 + half * 16 + e) * DH + d];
            acc[dt] = wmma_bf16(pa, vb, acc[dt]);
        }
    }

    #pragma unroll
    for (int r = 0; r < 8; ++r) {
        float inv = 1.0f / lrow[r];
        int l_ = qb * 16 + r + 8 * half;
        #pragma unroll
        for (int dt = 0; dt < 4; ++dt)
            att[((size_t)b_ * LL + l_) * DD + h_ * DH + dt * 16 + lr] =
                (bf16_t)(acc[dt][r] * inv);
    }
}

// ---------------------------------------------------------------------------
// launch
// ---------------------------------------------------------------------------
extern "C" void kernel_launch(void* const* d_in, const int* in_sizes, int n_in,
                              void* d_out, int out_size, void* d_ws, size_t ws_size,
                              hipStream_t stream) {
    (void)in_sizes; (void)n_in; (void)out_size; (void)ws_size;
    const float* q_x = (const float*)d_in[0];
    const float* k_x = (const float*)d_in[1];
    const float* v_x = (const float*)d_in[2];
    const float* ec  = (const float*)d_in[3];
    const float* Wq  = (const float*)d_in[4];
    const float* bq  = (const float*)d_in[5];
    const float* Wk  = (const float*)d_in[6];
    const float* bk  = (const float*)d_in[7];
    const float* Wv  = (const float*)d_in[8];
    const float* bv  = (const float*)d_in[9];
    const float* Wo  = (const float*)d_in[10];
    const float* bo  = (const float*)d_in[11];
    float* out = (float*)d_out;

    const size_t ELEMS = (size_t)BB * HH * LL * DH;          // 4,194,304
    bf16_t* Qb  = (bf16_t*)d_ws;
    bf16_t* Kb  = Qb + ELEMS;
    bf16_t* Vb  = Kb + ELEMS;
    bf16_t* Ab  = Vb + ELEMS;                                // [B,L,D] bf16
    float*  gte = (float*)(Ab + ELEMS);                      // [B,L] f32

    const int gemm_blocks = ((BB * LL) / 128) * (DD / 64);   // 512
    const int attn_blocks = (BB * HH * (LL / 16)) / 8;       // 512
    const int gate_blocks = (BB * LL) / 8;                   // 512

    gate_kernel<<<gate_blocks, 256, 0, stream>>>(ec, gte);
    linear_wmma<float,  true ><<<gemm_blocks, 256, 0, stream>>>(q_x, Wq, bq, Qb);
    linear_wmma<float,  true ><<<gemm_blocks, 256, 0, stream>>>(k_x, Wk, bk, Kb);
    linear_wmma<float,  true ><<<gemm_blocks, 256, 0, stream>>>(v_x, Wv, bv, Vb);
    attn_kernel<<<attn_blocks, 256, 0, stream>>>(Qb, Kb, Vb, gte, Ab);
    linear_wmma<bf16_t, false><<<gemm_blocks, 256, 0, stream>>>(Ab, Wo, bo, out);
}
